// ExtraPositionPromptSABottleneck_51204600103033
// MI455X (gfx1250) — matrix operand order repaired and checked by reading, not compiled, not run on hardware
//
#include <hip/hip_runtime.h>

// ===== problem constants =====
#define BATCH 32
#define DCH   512
#define CCH   256
#define NTOK  1024              // S*S
#define ROWS  (BATCH * NTOK)    // 32768

typedef __attribute__((ext_vector_type(16))) __bf16          bf16x16;
typedef __attribute__((ext_vector_type(8)))  float           f32x8;
typedef __attribute__((ext_vector_type(4)))  float           f32x4;
typedef __attribute__((ext_vector_type(16))) unsigned short  u16x16;
typedef __attribute__((ext_vector_type(8)))  unsigned short  u16x8;

__device__ __forceinline__ unsigned short f2bf(float f) {
  unsigned u = __builtin_bit_cast(unsigned, f);
  u += 0x7FFFu + ((u >> 16) & 1u);           // round-to-nearest-even
  return (unsigned short)(u >> 16);
}
__device__ __forceinline__ float silu(float y) { return y / (1.0f + __expf(-y)); }

__device__ __forceinline__ f32x8 wmma_bf16(bf16x16 a, bf16x16 b, f32x8 c) {
  return __builtin_amdgcn_wmma_f32_16x16x32_bf16(false, a, false, b, (short)0, c, false, false);
}

// A-fragment (16 rows x K=32) from row-major [rows][ld] bf16 (ushort view).
__device__ __forceinline__ bf16x16 load_a_frag(const unsigned short* base, int ld) {
  int lane = threadIdx.x & 31;
  int m = lane & 15, g = lane >> 4;
  const unsigned short* p = base + (size_t)m * ld;
  u16x8 lo = *(const u16x8*)(p + 8 * g);
  u16x8 hi = *(const u16x8*)(p + 16 + 8 * g);
  u16x16 r;
#pragma unroll
  for (int t = 0; t < 8; ++t) { r[t] = lo[t]; r[8 + t] = hi[t]; }
  return __builtin_bit_cast(bf16x16, r);
}

// B-fragment (K=32 x 16 cols) from B^T row-major [Ncols][ld] bf16.
__device__ __forceinline__ bf16x16 load_b_frag(const unsigned short* bt, int ld) {
  int lane = threadIdx.x & 31;
  int n = lane & 15, g = lane >> 4;
  const unsigned short* p = bt + (size_t)n * ld;
  u16x8 lo = *(const u16x8*)(p + 8 * g);
  u16x8 hi = *(const u16x8*)(p + 16 + 8 * g);
  u16x16 r;
#pragma unroll
  for (int t = 0; t < 8; ++t) { r[t] = lo[t]; r[8 + t] = hi[t]; }
  return __builtin_bit_cast(bf16x16, r);
}

// ===== weight pre-conversion: f32 -> bf16, done once (weights ~1MB total) =====
__global__ void k_cvt(const float* __restrict__ src, unsigned short* __restrict__ dst, int n) {
  int i = blockIdx.x * 256 + threadIdx.x;
  if (i < n) dst[i] = f2bf(src[i]);
}

// ===== pos[n][c] = rel_h[c, n>>5] + rel_w[c, n&31]  (bf16 [N][C]) =====
__global__ void k_pos(const float* __restrict__ rel_h, const float* __restrict__ rel_w,
                      unsigned short* __restrict__ pos) {
  int idx = blockIdx.x * 256 + threadIdx.x;
  int c = idx & (CCH - 1), n = idx >> 8;
  int h = n >> 5, w = n & 31;
  pos[(size_t)n * CCH + c] = f2bf(rel_h[c * 32 + h] + rel_w[c * 32 + w]);
}

// ===== xT[b][n][d] = bf16(x[b][d][n])  via LDS 32x32 tiles =====
__global__ void k_transpose(const float* __restrict__ x, unsigned short* __restrict__ xT) {
  __shared__ float tile[32][33];
  int b = blockIdx.z, d0 = blockIdx.y * 32, n0 = blockIdx.x * 32;
  int t = threadIdx.x;
  int j = t & 31, i0 = (t >> 5) * 4;
  const float* src = x + ((size_t)b * DCH + d0) * NTOK + n0;
#pragma unroll
  for (int r = 0; r < 4; ++r) tile[i0 + r][j] = src[(size_t)(i0 + r) * NTOK + j];
  __syncthreads();
  unsigned short* dst = xT + ((size_t)b * NTOK + n0) * DCH + d0;
  int i = t & 31, j0 = (t >> 5) * 4;
#pragma unroll
  for (int r = 0; r < 4; ++r) dst[(size_t)(j0 + r) * DCH + i] = f2bf(tile[i][j0 + r]);
}

// ===== pipelined 16x64 wave-tile GEMM body (A [rows][K], B^T [cols][K]) =====
// KTOT must be a multiple of 64. Ping-pong unroll-2: loop-carried buffers are
// written directly by loads (no rotation movs); prefetch overlaps the WMMAs.
template <int KTOT>
__device__ __forceinline__ void gemm_tile(const unsigned short* __restrict__ arow,
                                          const unsigned short* __restrict__ wrow,
                                          f32x8 acc[4]) {
  bf16x16 a0 = load_a_frag(arow, KTOT);
  bf16x16 b0[4];
#pragma unroll
  for (int t = 0; t < 4; ++t) b0[t] = load_b_frag(wrow + (size_t)(t * 16) * KTOT, KTOT);
  for (int k0 = 0; k0 < KTOT; k0 += 64) {
    const int k1 = k0 + 32;                             // always < KTOT
    bf16x16 a1 = load_a_frag(arow + k1, KTOT);
    bf16x16 b1[4];
#pragma unroll
    for (int t = 0; t < 4; ++t) b1[t] = load_b_frag(wrow + (size_t)(t * 16) * KTOT + k1, KTOT);
#pragma unroll
    for (int t = 0; t < 4; ++t) acc[t] = wmma_bf16(a0, b0[t], acc[t]);
    const int k2 = (k0 + 64 < KTOT) ? k0 + 64 : k1;     // last: benign cache-hit reload
    a0 = load_a_frag(arow + k2, KTOT);
#pragma unroll
    for (int t = 0; t < 4; ++t) b0[t] = load_b_frag(wrow + (size_t)(t * 16) * KTOT + k2, KTOT);
#pragma unroll
    for (int t = 0; t < 4; ++t) acc[t] = wmma_bf16(a1, b1[t], acc[t]);
  }
}

// ===== x1 = silu(bn(xT @ cv1_w^T))   [ROWS,512]x[512,256] -> bf16 [ROWS,256]
__global__ void __launch_bounds__(256) k_gemm_cv1(
    const unsigned short* __restrict__ xT, const unsigned short* __restrict__ w,
    const float* __restrict__ gamma, const float* __restrict__ beta,
    unsigned short* __restrict__ x1) {
  int gw = blockIdx.x * 8 + (threadIdx.x >> 5);
  int m0 = (gw >> 2) * 16, n0 = (gw & 3) * 64;
  int lane = threadIdx.x & 31, cl = lane & 15, g = lane >> 4;
  f32x8 acc[4] = {{0,0,0,0,0,0,0,0},{0,0,0,0,0,0,0,0},{0,0,0,0,0,0,0,0},{0,0,0,0,0,0,0,0}};
  gemm_tile<DCH>(xT + (size_t)m0 * DCH, w + (size_t)n0 * DCH, acc);
  const float inv = rsqrtf(1.0f + 1e-5f);
#pragma unroll
  for (int t = 0; t < 4; ++t) {
    int c = n0 + t * 16 + cl;
    float sc = gamma[c] * inv, be = beta[c];
#pragma unroll
    for (int r = 0; r < 8; ++r) {
      float y = silu(acc[t][r] * sc + be);
      x1[(size_t)(m0 + r + 8 * g) * CCH + c] = f2bf(y);
    }
  }
}

// ===== q/k/v/e projections (blockIdx.y selects); v stored transposed [B,C,N]
__global__ void __launch_bounds__(256) k_gemm_qkve(
    const unsigned short* __restrict__ x1,
    const unsigned short* __restrict__ qw, const float* __restrict__ qb2,
    const unsigned short* __restrict__ kw, const float* __restrict__ kb2,
    const unsigned short* __restrict__ vw, const float* __restrict__ vb2,
    const unsigned short* __restrict__ ew, const float* __restrict__ eb2,
    unsigned short* __restrict__ qo, unsigned short* __restrict__ ko,
    unsigned short* __restrict__ vT, unsigned short* __restrict__ eo) {
  int proj = blockIdx.y;
  const unsigned short* w = (proj == 0) ? qw : (proj == 1) ? kw : (proj == 2) ? vw : ew;
  const float* bias = (proj == 0) ? qb2 : (proj == 1) ? kb2 : (proj == 2) ? vb2 : eb2;
  int gw = blockIdx.x * 8 + (threadIdx.x >> 5);
  int m0 = (gw >> 2) * 16, n0 = (gw & 3) * 64;
  int lane = threadIdx.x & 31, cl = lane & 15, g = lane >> 4;
  f32x8 acc[4] = {{0,0,0,0,0,0,0,0},{0,0,0,0,0,0,0,0},{0,0,0,0,0,0,0,0},{0,0,0,0,0,0,0,0}};
  gemm_tile<CCH>(x1 + (size_t)m0 * CCH, w + (size_t)n0 * CCH, acc);
  unsigned short* out = (proj == 0) ? qo : (proj == 1) ? ko : eo;
#pragma unroll
  for (int t = 0; t < 4; ++t) {
    int c = n0 + t * 16 + cl;
    float be = bias[c];
#pragma unroll
    for (int r = 0; r < 8; ++r) {
      int row = m0 + r + 8 * g;
      unsigned short h = f2bf(acc[t][r] + be);
      if (proj == 2) vT[((size_t)(row >> 10) * CCH + c) * NTOK + (row & (NTOK - 1))] = h;
      else           out[(size_t)row * CCH + c] = h;
    }
  }
}

// ===== flash attention. Wave = 16 query rows; j-loop of 32; 48 WMMAs / iter.
__global__ void __launch_bounds__(256) k_attn(
    const unsigned short* __restrict__ q, const unsigned short* __restrict__ k,
    const unsigned short* __restrict__ vT, const unsigned short* __restrict__ e,
    const unsigned short* __restrict__ pos, unsigned short* __restrict__ out) {
  __shared__ unsigned short lds_p[8][16 * 32];       // per-wave P staging (D-frag -> A-frag)
  int wv = threadIdx.x >> 5;
  int gw = blockIdx.x * 8 + wv;
  int i0 = gw * 16;                                  // global row in [0,32768)
  int b  = i0 >> 10;
  int lane = threadIdx.x & 31, cl = lane & 15, g = lane >> 4;

  const unsigned short* qbase = q   + (size_t)i0 * CCH;
  const unsigned short* pbase = pos + (size_t)(i0 & (NTOK - 1)) * CCH;
  const unsigned short* kbase = k   + (size_t)b * NTOK * CCH;
  const unsigned short* ebase = e   + (size_t)b * NTOK * CCH;
  const unsigned short* vbase = vT  + (size_t)b * CCH * NTOK;

  bf16x16 qf[8], pf[8];
#pragma unroll
  for (int cc = 0; cc < 8; ++cc) {
    qf[cc] = load_a_frag(qbase + cc * 32, CCH);
    pf[cc] = load_a_frag(pbase + cc * 32, CCH);
  }

  f32x8 O[16];
#pragma unroll
  for (int ct = 0; ct < 16; ++ct) O[ct] = (f32x8){0,0,0,0,0,0,0,0};
  float m[8], l[8];
#pragma unroll
  for (int r = 0; r < 8; ++r) { m[r] = -3.0e38f; l[r] = 0.0f; }

  for (int j0 = 0; j0 < NTOK; j0 += 32) {
    f32x8 s0 = (f32x8){0,0,0,0,0,0,0,0};
    f32x8 s1 = (f32x8){0,0,0,0,0,0,0,0};
    // ---- S = q.k + pos.e over C=256, software-pipelined over 32-wide c-chunks ----
    bf16x16 kb0 = load_b_frag(kbase + (size_t)j0 * CCH, CCH);
    bf16x16 eb0 = load_b_frag(ebase + (size_t)j0 * CCH, CCH);
    bf16x16 kb1 = load_b_frag(kbase + (size_t)(j0 + 16) * CCH, CCH);
    bf16x16 eb1 = load_b_frag(ebase + (size_t)(j0 + 16) * CCH, CCH);
#pragma unroll
    for (int cc = 0; cc < 8; ++cc) {
      int cn = (cc < 7) ? (cc + 1) * 32 : cc * 32;   // last iter: benign reload
      bf16x16 nk0 = load_b_frag(kbase + (size_t)j0 * CCH + cn, CCH);
      bf16x16 ne0 = load_b_frag(ebase + (size_t)j0 * CCH + cn, CCH);
      bf16x16 nk1 = load_b_frag(kbase + (size_t)(j0 + 16) * CCH + cn, CCH);
      bf16x16 ne1 = load_b_frag(ebase + (size_t)(j0 + 16) * CCH + cn, CCH);
      s0 = wmma_bf16(qf[cc], kb0, s0);
      s0 = wmma_bf16(pf[cc], eb0, s0);
      s1 = wmma_bf16(qf[cc], kb1, s1);
      s1 = wmma_bf16(pf[cc], eb1, s1);
      kb0 = nk0; eb0 = ne0; kb1 = nk1; eb1 = ne1;
    }
    // ---- online softmax: D-frag elem r <-> row (r + 8g); cols across 16-lane half ----
#pragma unroll
    for (int r = 0; r < 8; ++r) {
      float mx = fmaxf(s0[r], s1[r]);
#pragma unroll
      for (int off = 1; off < 16; off <<= 1) mx = fmaxf(mx, __shfl_xor(mx, off, 32));
      float mn = fmaxf(m[r], mx);
      float sc = __expf(m[r] - mn);
      m[r] = mn;
      l[r] *= sc;
#pragma unroll
      for (int ct = 0; ct < 16; ++ct) O[ct][r] *= sc;
      float p0 = __expf(s0[r] - mn);
      float p1 = __expf(s1[r] - mn);
      float rs = p0 + p1;
#pragma unroll
      for (int off = 1; off < 16; off <<= 1) rs += __shfl_xor(rs, off, 32);
      l[r] += rs;
      int row = r + 8 * g;
      lds_p[wv][row * 32 + cl]      = f2bf(p0);    // K = local j 0..15
      lds_p[wv][row * 32 + 16 + cl] = f2bf(p1);    // K = local j 16..31
    }
    asm volatile("s_wait_dscnt 0" ::: "memory");   // wave-private LDS RAW
    bf16x16 pa = load_a_frag(&lds_p[wv][0], 32);
    // ---- O += P @ v over 16 c-tiles, software-pipelined ----
    bf16x16 vb = load_b_frag(vbase + j0, NTOK);
#pragma unroll
    for (int ct = 0; ct < 16; ++ct) {
      int cn = (ct < 15) ? ct + 1 : ct;            // last iter: benign reload
      bf16x16 vn = load_b_frag(vbase + (size_t)(cn * 16) * NTOK + j0, NTOK);
      O[ct] = wmma_bf16(pa, vb, O[ct]);
      vb = vn;
    }
  }
#pragma unroll
  for (int r = 0; r < 8; ++r) {
    float inv = 1.0f / l[r];
    size_t row = (size_t)(i0 + r + 8 * g);
#pragma unroll
    for (int ct = 0; ct < 16; ++ct)
      out[row * CCH + ct * 16 + cl] = f2bf(O[ct][r] * inv);
  }
}

// ===== y = x + silu(bn(attn_out @ cv2_w^T)); coalesced float4 stores
__global__ void __launch_bounds__(256) k_gemm_cv2(
    const unsigned short* __restrict__ ao, const unsigned short* __restrict__ w,
    const float* __restrict__ gamma, const float* __restrict__ beta,
    const float* __restrict__ x, float* __restrict__ out) {
  int gw = blockIdx.x * 8 + (threadIdx.x >> 5);
  int m0 = (gw >> 3) * 16, n0 = (gw & 7) * 64;
  int lane = threadIdx.x & 31, cl = lane & 15, g = lane >> 4;
  f32x8 acc[4] = {{0,0,0,0,0,0,0,0},{0,0,0,0,0,0,0,0},{0,0,0,0,0,0,0,0},{0,0,0,0,0,0,0,0}};
  gemm_tile<CCH>(ao + (size_t)m0 * CCH, w + (size_t)n0 * CCH, acc);
  const float inv = rsqrtf(1.0f + 1e-5f);
  int bb = m0 >> 10;
  int nl = (m0 & (NTOK - 1)) + 8 * g;                 // rows r -> consecutive n
#pragma unroll
  for (int t = 0; t < 4; ++t) {
    int d = n0 + t * 16 + cl;
    float sc = gamma[d] * inv, be = beta[d];
    size_t base = ((size_t)bb * DCH + d) * NTOK + nl;
    f32x4 xa = *(const f32x4*)(x + base);
    f32x4 xb = *(const f32x4*)(x + base + 4);
    f32x4 y0, y1;
#pragma unroll
    for (int r = 0; r < 4; ++r) {
      y0[r] = xa[r] + silu(acc[t][r] * sc + be);
      y1[r] = xb[r] + silu(acc[t][4 + r] * sc + be);
    }
    *(f32x4*)(out + base)     = y0;
    *(f32x4*)(out + base + 4) = y1;
  }
}

extern "C" void kernel_launch(void* const* d_in, const int* in_sizes, int n_in,
                              void* d_out, int out_size, void* d_ws, size_t ws_size,
                              hipStream_t stream) {
  const float* x     = (const float*)d_in[0];
  const float* cv1_w = (const float*)d_in[1];
  const float* cv1_g = (const float*)d_in[2];
  const float* cv1_b = (const float*)d_in[3];
  const float* q_w   = (const float*)d_in[4];
  const float* q_b   = (const float*)d_in[5];
  const float* k_w   = (const float*)d_in[6];
  const float* k_b   = (const float*)d_in[7];
  const float* v_w   = (const float*)d_in[8];
  const float* v_b   = (const float*)d_in[9];
  const float* e_w   = (const float*)d_in[10];
  const float* e_b   = (const float*)d_in[11];
  const float* rel_h = (const float*)d_in[12];
  const float* rel_w = (const float*)d_in[13];
  const float* cv2_w = (const float*)d_in[14];
  const float* cv2_g = (const float*)d_in[15];
  const float* cv2_b = (const float*)d_in[16];

  // workspace layout (bytes); ~119.1 MB total. attn_out reuses xT (dead after cv1).
  char* ws = (char*)d_ws;
  unsigned short* xT   = (unsigned short*)(ws);              // 32 MB   [B,N,D] bf16
  unsigned short* x1   = (unsigned short*)(ws + 33554432);   // 16 MB   [B,N,C]
  unsigned short* qbuf = (unsigned short*)(ws + 50331648);   // 16 MB
  unsigned short* kbuf = (unsigned short*)(ws + 67108864);   // 16 MB
  unsigned short* vbuf = (unsigned short*)(ws + 83886080);   // 16 MB   [B,C,N]
  unsigned short* ebuf = (unsigned short*)(ws + 100663296);  // 16 MB
  unsigned short* pos  = (unsigned short*)(ws + 117440512);  // 512 KB  [N,C]
  unsigned short* w1b  = (unsigned short*)(ws + 117964800);  // 256 KB  cv1_w bf16
  unsigned short* wqb  = (unsigned short*)(ws + 118226944);  // 128 KB
  unsigned short* wkb  = (unsigned short*)(ws + 118358016);  // 128 KB
  unsigned short* wvb  = (unsigned short*)(ws + 118489088);  // 128 KB
  unsigned short* web  = (unsigned short*)(ws + 118620160);  // 128 KB
  unsigned short* w2b  = (unsigned short*)(ws + 118751232);  // 256 KB  cv2_w bf16
  unsigned short* aout = xT;                                 // reuse

  // --- weight pre-conversion (one-time, ~1MB) ---
  k_cvt<<<(CCH * DCH) / 256, 256, 0, stream>>>(cv1_w, w1b, CCH * DCH);
  k_cvt<<<(CCH * CCH) / 256, 256, 0, stream>>>(q_w, wqb, CCH * CCH);
  k_cvt<<<(CCH * CCH) / 256, 256, 0, stream>>>(k_w, wkb, CCH * CCH);
  k_cvt<<<(CCH * CCH) / 256, 256, 0, stream>>>(v_w, wvb, CCH * CCH);
  k_cvt<<<(CCH * CCH) / 256, 256, 0, stream>>>(e_w, web, CCH * CCH);
  k_cvt<<<(DCH * CCH) / 256, 256, 0, stream>>>(cv2_w, w2b, DCH * CCH);

  k_pos<<<(NTOK * CCH) / 256, 256, 0, stream>>>(rel_h, rel_w, pos);
  k_transpose<<<dim3(NTOK / 32, DCH / 32, BATCH), 256, 0, stream>>>(x, xT);
  k_gemm_cv1<<<(ROWS / 16) * (CCH / 64) / 8, 256, 0, stream>>>(xT, w1b, cv1_g, cv1_b, x1);
  k_gemm_qkve<<<dim3((ROWS / 16) * (CCH / 64) / 8, 4), 256, 0, stream>>>(
      x1, wqb, q_b, wkb, k_b, wvb, v_b, web, e_b, qbuf, kbuf, vbuf, ebuf);
  k_attn<<<(ROWS / 16) / 8, 256, 0, stream>>>(qbuf, kbuf, vbuf, ebuf, pos, aout);
  k_gemm_cv2<<<(ROWS / 16) * (DCH / 64) / 8, 256, 0, stream>>>(
      aout, w2b, cv2_g, cv2_b, x, (float*)d_out);
}
